// CFE_81475529605505
// MI455X (gfx1250) — matrix operand order repaired and checked by reading, not compiled, not run on hardware
//
#include <hip/hip_runtime.h>
#include <hip/hip_bf16.h>

// ---------------------------------------------------------------------------
// Types for the CDNA5 WMMA builtins / TDM descriptors
// ---------------------------------------------------------------------------
typedef __bf16  v16bf __attribute__((ext_vector_type(16)));
typedef float   v8f   __attribute__((ext_vector_type(8)));
typedef unsigned int v4u __attribute__((ext_vector_type(4)));
typedef unsigned int v8u __attribute__((ext_vector_type(8)));

#define MVOX   100000
#define TILE_M 128
#define NTHREADS 256

// float -> bf16 (round to nearest even), storage-only use of __bf16
__device__ __forceinline__ __bf16 f2bf(float f) {
    union { float f; unsigned u; } a; a.f = f;
    unsigned u = a.u;
    unsigned r = u + 0x7FFFu + ((u >> 16) & 1u);
    union { unsigned short s; __bf16 b; } o;
    o.s = (unsigned short)(r >> 16);
    return o.b;
}

// ---------------------------------------------------------------------------
// Prep kernels: f32 -> bf16 conversion (features, and weights transposed)
// ---------------------------------------------------------------------------
__global__ void cvt_weight_kernel(const float* __restrict__ in, __bf16* __restrict__ out,
                                  int KOFF, int CIN, int CINP, int COUT) {
    int idx = blockIdx.x * blockDim.x + threadIdx.x;
    int total = KOFF * COUT * CINP;
    if (idx >= total) return;
    int c  = idx % CINP;
    int n  = (idx / CINP) % COUT;
    int ko = idx / (CINP * COUT);
    float v = (c < CIN) ? in[((size_t)ko * CIN + c) * COUT + n] : 0.0f;
    out[idx] = f2bf(v);
}

__global__ void cvt_feat_kernel(const float* __restrict__ in, __bf16* __restrict__ out,
                                int M, int CIN, int CINP) {
    int idx = blockIdx.x * blockDim.x + threadIdx.x;
    int total = M * CINP;
    if (idx >= total) return;
    int c = idx % CINP;
    int m = idx / CINP;
    float v = (c < CIN) ? in[(size_t)m * CIN + c] : 0.0f;
    out[idx] = f2bf(v);
}

// ---------------------------------------------------------------------------
// Fused gather + GEMM (sparse 3^3 conv when nbr!=null, dense 1x1 when null)
//   in   : [M, CIN]  bf16 row-major
//   Wt   : [koff, COUT, CIN] bf16 (weight TRANSPOSED: row = output channel)
// Block = 256 threads = 8 waves; block computes a 128-row x COUT tile.
// A tile: per-lane masked gather (vmem -> LDS). B tile: TDM tensor_load_to_lds.
// ---------------------------------------------------------------------------
template<int CIN, int COUT>
__launch_bounds__(NTHREADS)
__global__ void spconv_wmma(const __bf16* __restrict__ in,
                            const __bf16* __restrict__ Wt,
                            const float*  __restrict__ bias,
                            const int*    __restrict__ nbr,   // [koff, M] or null
                            int koff, int M,
                            const float*  __restrict__ film,  // [M, 2*COUT] or null
                            const float*  __restrict__ resid, // [M, COUT] or null
                            float*        __restrict__ outF,  // [M, COUT] or null
                            __bf16*       __restrict__ outB,  // [M, COUT] or null
                            int doRelu) {
    constexpr int KCH = (CIN < 64) ? CIN : 64;    // K chunk staged in LDS
    constexpr int NT  = COUT / 16;                // wmma col tiles per wave
    constexpr int QPR = KCH / 8;                  // uint4 (8 bf16) per staged row
    constexpr int QPT = QPR / 2;                  // quads per thread (2 thr/row)

    __shared__ __bf16 sA[TILE_M * KCH];
    __shared__ __bf16 sB[COUT * KCH];

    const int tid  = threadIdx.x;
    const int wv   = tid >> 5;
    const int lane = tid & 31;
    const int l15  = lane & 15;
    const int lhalf = lane >> 4;
    const int wrow = wv * 16;
    const int m0   = blockIdx.x * TILE_M;

    v8f acc[NT];
#pragma unroll
    for (int i = 0; i < NT; ++i) acc[i] = v8f{0.f,0.f,0.f,0.f,0.f,0.f,0.f,0.f};

    // staging identity (per-thread constants)
    const int rStg  = tid >> 1;        // row this thread stages (2 threads/row)
    const int pStg  = tid & 1;
    const unsigned ldsB = (unsigned)(size_t)(&sB[0]);   // LDS byte offset of sB

    for (int ko = 0; ko < koff; ++ko) {
        // prefetch next weight slab toward L2 while we work on this one
        if (ko + 1 < koff && tid == 0)
            __builtin_prefetch(Wt + ((size_t)(ko + 1) * COUT) * CIN, 0, 1);

        // resolve source row once per k-offset
        int mRow = m0 + rStg;
        int src;
        if (nbr) src = (mRow < M) ? nbr[(size_t)ko * M + mRow] : -1;
        else     src = (mRow < M) ? mRow : -1;

        for (int kc = 0; kc < CIN / KCH; ++kc) {
            // ---- stage B via Tensor Data Mover: COUT x KCH tile of Wt[ko] ----
            // D# group0: count=1 | lds_addr | global_addr(57b) | type=2
            // D# group1: data_size=2B | tensor=[CIN x COUT] | tile=[KCH x COUT]
            //            | dim0_stride=CIN
            if (wv == 0) {
                unsigned long long gaddr =
                    (unsigned long long)(size_t)(Wt + ((size_t)ko * COUT) * CIN + kc * KCH);
                union { v4u v; unsigned u[4]; } g0;
                g0.u[0] = 1u;                                   // count=1, user mode
                g0.u[1] = ldsB;                                 // lds_addr
                g0.u[2] = (unsigned)(gaddr & 0xFFFFFFFFull);    // global_addr[31:0]
                g0.u[3] = (unsigned)((gaddr >> 32) & 0x01FFFFFFull) | (2u << 30); // type=2
                union { v8u v; unsigned u[8]; } g1;
                g1.u[0] = (1u << 16);                           // data_size=1 (2 bytes)
                g1.u[1] = ((unsigned)CIN & 0xFFFFu) << 16;      // tensor_dim0 lo16
                g1.u[2] = ((unsigned)CIN >> 16) | (((unsigned)COUT & 0xFFFFu) << 16); // dim0 hi | dim1 lo
                g1.u[3] = ((unsigned)COUT >> 16) | ((unsigned)KCH << 16);            // dim1 hi | tile_dim0
                g1.u[4] = (unsigned)COUT;                       // tile_dim1 (tile_dim2=0)
                g1.u[5] = (unsigned)CIN;                        // tensor_dim0_stride lo32
                g1.u[6] = 0u;                                   // stride hi | dim1_stride lo
                g1.u[7] = 0u;
                asm volatile("tensor_load_to_lds %0, %1"
                             :: "s"(g0.v), "s"(g1.v)
                             : "memory");
            }

            // ---- stage A: 128 x KCH gathered rows (masked, per-lane vmem) ----
            {
                const uint4* gsrc = (src >= 0)
                    ? (const uint4*)(in + (size_t)src * CIN + kc * KCH) : nullptr;
                uint4* ldst = (uint4*)(&sA[rStg * KCH]);
#pragma unroll
                for (int j = 0; j < QPT; ++j) {
                    int qi = pStg * QPT + j;
                    uint4 v;
                    if (gsrc) v = gsrc[qi];
                    else { v.x = 0u; v.y = 0u; v.z = 0u; v.w = 0u; }
                    ldst[qi] = v;
                }
            }

            // TDM completion (wave 0) before the workgroup barrier releases readers
            if (wv == 0) __builtin_amdgcn_s_wait_tensorcnt(0);
            __syncthreads();

            // ---- compute: wmma over this K chunk ----
#pragma unroll
            for (int kk = 0; kk < KCH; kk += 32) {
                union { v16bf v; uint4 q[2]; } a;
                const __bf16* ap = &sA[(wrow + l15) * KCH + kk + lhalf * 8];
                a.q[0] = *(const uint4*)(ap);        // K = kk + h*8 .. +7
                a.q[1] = *(const uint4*)(ap + 16);   // K = kk+16+h*8 .. +7
#pragma unroll
                for (int nt = 0; nt < NT; ++nt) {
                    union { v16bf v; uint4 q[2]; } b;
                    const __bf16* bp = &sB[(nt * 16 + l15) * KCH + kk + lhalf * 8];
                    b.q[0] = *(const uint4*)(bp);
                    b.q[1] = *(const uint4*)(bp + 16);
                    acc[nt] = __builtin_amdgcn_wmma_f32_16x16x32_bf16(
                        false, a.v, false, b.v, (short)0, acc[nt], false, false);
                }
            }
            __syncthreads();
        }
    }

    // ---- epilogue: bias / relu / FiLM / residual; f32 and/or bf16 stores ----
#pragma unroll
    for (int nt = 0; nt < NT; ++nt) {
        int n = nt * 16 + l15;
        float bval = bias[n];
#pragma unroll
        for (int r = 0; r < 8; ++r) {
            int row = m0 + wrow + lhalf * 8 + r;
            if (row < M) {
                float v = acc[nt][r] + bval;
                if (doRelu) v = fmaxf(v, 0.0f);
                if (film) {
                    float beta  = film[(size_t)row * (2 * COUT) + n];
                    float gamma = film[(size_t)row * (2 * COUT) + COUT + n];
                    v = v * beta + gamma;
                }
                if (resid) v += resid[(size_t)row * COUT + n];
                if (outF)  outF[(size_t)row * COUT + n] = v;
                if (outB)  outB[(size_t)row * COUT + n] = f2bf(v);
            }
        }
    }
}

// ---------------------------------------------------------------------------
// Host: workspace layout + launch sequence
// ---------------------------------------------------------------------------
static inline size_t ws_take(size_t& off, size_t bytes) {
    size_t o = off;
    off = (off + bytes + 255) & ~(size_t)255;
    return o;
}

extern "C" void kernel_launch(void* const* d_in, const int* in_sizes, int n_in,
                              void* d_out, int out_size, void* d_ws, size_t ws_size,
                              hipStream_t stream) {
    (void)in_sizes; (void)n_in; (void)out_size; (void)ws_size;
    const int M = MVOX;

    const float* x    = (const float*)d_in[0];
    const float* cond = (const float*)d_in[1];
    const float* W1a  = (const float*)d_in[2];  const float* b1a = (const float*)d_in[3];
    const float* W1b  = (const float*)d_in[4];  const float* b1b = (const float*)d_in[5];
    const float* W2a  = (const float*)d_in[6];  const float* b2a = (const float*)d_in[7];
    const float* W2b  = (const float*)d_in[8];  const float* b2b = (const float*)d_in[9];
    const float* WQ1  = (const float*)d_in[10]; const float* bQ1 = (const float*)d_in[11];
    const float* WQ2  = (const float*)d_in[12]; const float* bQ2 = (const float*)d_in[13];
    const float* WQ3  = (const float*)d_in[14]; const float* bQ3 = (const float*)d_in[15];
    const float* WQ4  = (const float*)d_in[16]; const float* bQ4 = (const float*)d_in[17];
    const int*   nbr  = (const int*)d_in[18];

    char* ws = (char*)d_ws;
    size_t off = 0;
    // bf16 transposed weights
    __bf16* W1aT = (__bf16*)(ws + ws_take(off, (size_t)27*128*128*2));
    __bf16* W1bT = (__bf16*)(ws + ws_take(off, (size_t)27*128*128*2));
    __bf16* W2aT = (__bf16*)(ws + ws_take(off, (size_t)27*128*128*2));
    __bf16* W2bT = (__bf16*)(ws + ws_take(off, (size_t)27*128*128*2));
    __bf16* WQ1T = (__bf16*)(ws + ws_take(off, (size_t)27*64*32*2));
    __bf16* WQ2T = (__bf16*)(ws + ws_take(off, (size_t)128*64*2));
    __bf16* WQ3T = (__bf16*)(ws + ws_take(off, (size_t)256*128*2));
    __bf16* WQ4T = (__bf16*)(ws + ws_take(off, (size_t)256*256*2));
    // bf16 features
    __bf16* xB    = (__bf16*)(ws + ws_take(off, (size_t)M*128*2));
    __bf16* condB = (__bf16*)(ws + ws_take(off, (size_t)M*32*2));
    __bf16* q1B   = (__bf16*)(ws + ws_take(off, (size_t)M*64*2));
    __bf16* q2B   = (__bf16*)(ws + ws_take(off, (size_t)M*128*2));
    __bf16* q3B   = (__bf16*)(ws + ws_take(off, (size_t)M*256*2));
    float*  qF    = (float*) (ws + ws_take(off, (size_t)M*256*4));
    // h buffers reuse dead q regions: q3B (dead after qd) and q2B (dead after qc)
    __bf16* h1B = q3B;                      // M*128 bf16 fits in M*256 region
    __bf16* h2B = q3B + (size_t)M * 128;    // second half of q3B region
    __bf16* h3B = q2B;                      // M*128 bf16

    auto g1 = [](size_t total) { return dim3((unsigned)((total + 255) / 256)); };
    const dim3 blk(256);
    const dim3 gM((M + TILE_M - 1) / TILE_M);

    // ---- prep: weight + feature conversion ----
    cvt_weight_kernel<<<g1((size_t)27*128*128), blk, 0, stream>>>(W1a, W1aT, 27, 128, 128, 128);
    cvt_weight_kernel<<<g1((size_t)27*128*128), blk, 0, stream>>>(W1b, W1bT, 27, 128, 128, 128);
    cvt_weight_kernel<<<g1((size_t)27*128*128), blk, 0, stream>>>(W2a, W2aT, 27, 128, 128, 128);
    cvt_weight_kernel<<<g1((size_t)27*128*128), blk, 0, stream>>>(W2b, W2bT, 27, 128, 128, 128);
    cvt_weight_kernel<<<g1((size_t)27*64*32),   blk, 0, stream>>>(WQ1, WQ1T, 27, 16, 32, 64);
    cvt_weight_kernel<<<g1((size_t)128*64),     blk, 0, stream>>>(WQ2, WQ2T, 1, 64, 64, 128);
    cvt_weight_kernel<<<g1((size_t)256*128),    blk, 0, stream>>>(WQ3, WQ3T, 1, 128, 128, 256);
    cvt_weight_kernel<<<g1((size_t)256*256),    blk, 0, stream>>>(WQ4, WQ4T, 1, 256, 256, 256);
    cvt_feat_kernel<<<g1((size_t)M*128), blk, 0, stream>>>(x, xB, M, 128, 128);
    cvt_feat_kernel<<<g1((size_t)M*32),  blk, 0, stream>>>(cond, condB, M, 16, 32);

    // ---- conditional stream q (needed before FiLM) ----
    spconv_wmma<32, 64><<<gM, blk, 0, stream>>>(condB, WQ1T, bQ1, nbr, 27, M,
                                                nullptr, nullptr, nullptr, q1B, 1);
    spconv_wmma<64, 128><<<gM, blk, 0, stream>>>(q1B, WQ2T, bQ2, nullptr, 1, M,
                                                 nullptr, nullptr, nullptr, q2B, 1);
    spconv_wmma<128, 256><<<gM, blk, 0, stream>>>(q2B, WQ3T, bQ3, nullptr, 1, M,
                                                  nullptr, nullptr, nullptr, q3B, 1);
    spconv_wmma<256, 256><<<gM, blk, 0, stream>>>(q3B, WQ4T, bQ4, nullptr, 1, M,
                                                  nullptr, nullptr, qF, nullptr, 0);

    // ---- main stream: conv_1 (relu, then FiLM fused), conv_2 (+residual) ----
    spconv_wmma<128, 128><<<gM, blk, 0, stream>>>(xB, W1aT, b1a, nbr, 27, M,
                                                  nullptr, nullptr, nullptr, h1B, 1);
    spconv_wmma<128, 128><<<gM, blk, 0, stream>>>(h1B, W1bT, b1b, nbr, 27, M,
                                                  qF /*FiLM*/, nullptr, nullptr, h2B, 0);
    spconv_wmma<128, 128><<<gM, blk, 0, stream>>>(h2B, W2aT, b2a, nbr, 27, M,
                                                  nullptr, nullptr, nullptr, h3B, 1);
    spconv_wmma<128, 128><<<gM, blk, 0, stream>>>(h3B, W2bT, b2b, nbr, 27, M,
                                                  nullptr, x /*residual*/, (float*)d_out, nullptr, 0);
}
